// SynthesisFusedBlock_26852135535108
// MI455X (gfx1250) — matrix operand order repaired and checked by reading, not compiled, not run on hardware
//
#include <hip/hip_runtime.h>
#include <hip/hip_bf16.h>
#include <math.h>

// ---------------------------------------------------------------------------
// SynthesisFusedBlock for MI455X (gfx1250, wave32, WMMA)
// Heavy convs (expand 3x3, proj 1x1, GDN 1x1) run as implicit GEMMs on
// v_wmma_f32_16x16x32_bf16. Wave tile = 32M x 96N (expand) / 32M x 48N
// (proj/GDN): each A fragment reused across 6/3 B fragments and each B
// fragment reused across 2 M subtiles -> ~1.3 fragment loads per WMMA.
// A 32-wide M tile sits inside one image row, so the vertical halo check
// and row base address are wave-uniform scalars.
// ---------------------------------------------------------------------------

typedef __bf16 bf16_t;
typedef __attribute__((ext_vector_type(16))) __bf16 v16bf;
typedef __attribute__((ext_vector_type(8)))  __bf16 v8bf;
typedef __attribute__((ext_vector_type(8)))  float  v8f;

#define BN 4
#define CH 192
#define HH 64          // half-res spatial
#define H2 128         // full-res spatial
#define W2 128
#define EX 768         // expansion channels
#define K9 (CH * 9)    // 1728, K of expand GEMM
#define NPIX (BN * H2 * W2)  // 65536 output pixels

// CDF 9/7 synthesis filters
__device__ __constant__ float G0c[7] = {
    -0.091271763114f, -0.057543526228f, 0.591271763114f, 1.115087052457f,
     0.591271763114f, -0.057543526228f, -0.091271763114f};
__device__ __constant__ float G1c[9] = {
     0.026748757411f, 0.016864118443f, -0.078223266529f, -0.266864118443f,
     0.602949018236f, -0.266864118443f, -0.078223266529f, 0.016864118443f,
     0.026748757411f};

// ------------------------- weight repack (f32 -> bf16 [N][K]) --------------

__global__ void prep_wexp(const float* __restrict__ w, bf16_t* __restrict__ o) {
  int id = blockIdx.x * blockDim.x + threadIdx.x;
  if (id >= EX * CH * 9) return;
  // w[e][c][ky][kx] flat; K ordering = tap*CH + c so one 32-wide K-block
  // stays inside a single (ky,kx) tap (192 % 32 == 0).
  int e = id / (CH * 9);
  int c = (id / 9) % CH;
  int tap = id % 9;
  o[(size_t)e * K9 + tap * CH + c] = (bf16_t)w[id];
}

__global__ void prep_copy_bf16(const float* __restrict__ w, bf16_t* __restrict__ o, int n) {
  int id = blockIdx.x * blockDim.x + threadIdx.x;
  if (id < n) o[id] = (bf16_t)w[id];
}

// ------------------------- mask union (half-res) ---------------------------

__global__ void mu_kernel(const int* __restrict__ llm, const int* __restrict__ m1,
                          const int* __restrict__ m2, const int* __restrict__ m3,
                          float* __restrict__ mu) {
  int id = blockIdx.x * blockDim.x + threadIdx.x;
  if (id >= BN * CH * HH * HH) return;
  int v = llm[id];
  v = max(v, m1[id]); v = max(v, m2[id]); v = max(v, m3[id]);
  mu[id] = (float)v;
}

// ------------------------- IDWT vertical pass ------------------------------

__global__ void idwt_vert(const float* __restrict__ ll, const float* __restrict__ b1,
                          const float* __restrict__ b2, const float* __restrict__ b3,
                          const int* __restrict__ llm, const int* __restrict__ m1,
                          const int* __restrict__ m2, const int* __restrict__ m3,
                          float* __restrict__ lo, float* __restrict__ hi) {
  int id = blockIdx.x * blockDim.x + threadIdx.x;
  if (id >= BN * CH * H2 * HH) return;
  int x  = id % HH;
  int y  = (id / HH) % H2;
  int bc = id / (HH * H2);
  int base = bc * (HH * HH);
  float aLo = 0.f, aHi = 0.f;
#pragma unroll
  for (int j = 0; j < 7; ++j) {
    int t = y + j - 3;
    if ((t & 1) == 0) {
      int r = t >> 1;
      if (r >= 0 && r < HH) {
        int idx = base + r * HH + x;
        aLo += G0c[j] * ll[idx] * (float)llm[idx];
        aHi += G0c[j] * b2[idx] * (float)m2[idx];
      }
    }
  }
#pragma unroll
  for (int j = 0; j < 9; ++j) {
    int t = y + j - 4;
    if ((t & 1) == 0) {
      int r = t >> 1;
      if (r >= 0 && r < HH) {
        int idx = base + r * HH + x;
        aLo += G1c[j] * b1[idx] * (float)m1[idx];
        aHi += G1c[j] * b3[idx] * (float)m3[idx];
      }
    }
  }
  lo[id] = aLo;
  hi[id] = aHi;
}

// ------------------------- IDWT horizontal pass ----------------------------

__global__ void idwt_horiz(const float* __restrict__ lo, const float* __restrict__ hi,
                           float* __restrict__ pre) {
  int id = blockIdx.x * blockDim.x + threadIdx.x;
  if (id >= BN * CH * H2 * W2) return;
  int x  = id % W2;
  int yb = id / W2;                 // (bc*128 + y)
  int base = yb * HH;
  float acc = 0.f;
#pragma unroll
  for (int j = 0; j < 7; ++j) {
    int t = x + j - 3;
    if ((t & 1) == 0) {
      int r = t >> 1;
      if (r >= 0 && r < HH) acc += G0c[j] * lo[base + r];
    }
  }
#pragma unroll
  for (int j = 0; j < 9; ++j) {
    int t = x + j - 4;
    if ((t & 1) == 0) {
      int r = t >> 1;
      if (r >= 0 && r < HH) acc += G1c[j] * hi[base + r];
    }
  }
  pre[id] = acc;
}

// ------------------------- depthwise 3x3 partial conv ----------------------

__global__ void cast_pconv(const float* __restrict__ pre, const float* __restrict__ mu,
                           const float* __restrict__ wc, const float* __restrict__ bc,
                           bf16_t* __restrict__ xm, bf16_t* __restrict__ mcast) {
  int id = blockIdx.x * blockDim.x + threadIdx.x;
  if (id >= BN * H2 * W2 * CH) return;
  int c = id % CH;
  int x = (id / CH) % W2;
  int y = (id / (CH * W2)) % H2;
  int b = id / (CH * W2 * H2);
  float msum = 0.f, conv = 0.f;
#pragma unroll
  for (int ky = 0; ky < 3; ++ky) {
#pragma unroll
    for (int kx = 0; kx < 3; ++kx) {
      int sy = y + ky - 1, sx = x + kx - 1;
      if (sy >= 0 && sy < H2 && sx >= 0 && sx < W2) {
        float m = mu[((b * CH + c) * HH + (sy >> 1)) * HH + (sx >> 1)];
        msum += m;
        conv += m * pre[((b * CH + c) * H2 + sy) * W2 + sx] * wc[c * 9 + ky * 3 + kx];
      }
    }
  }
  float valid = (msum > 0.f) ? 1.f : 0.f;
  float outv = ((msum > 0.f) ? conv * (9.f / fmaxf(msum, 1.f)) : 0.f) + bc[c];
  xm[id]    = (bf16_t)(outv * valid);
  mcast[id] = (bf16_t)valid;
}

// ------------------------- expand-conv renorm plumbing ---------------------

__global__ void chansum(const bf16_t* __restrict__ mcast, float* __restrict__ s) {
  int p = blockIdx.x * blockDim.x + threadIdx.x;
  if (p >= NPIX) return;
  const bf16_t* row = mcast + (size_t)p * CH;
  float acc = 0.f;
  for (int c = 0; c < CH; ++c) acc += (float)row[c];
  s[p] = acc;
}

__global__ void renorm_expand(const float* __restrict__ s, float* __restrict__ r,
                              float* __restrict__ mh) {
  int p = blockIdx.x * blockDim.x + threadIdx.x;
  if (p >= NPIX) return;
  int b = p >> 14, rem = p & 16383;
  int y = rem >> 7, x = rem & 127;
  float msum = 0.f;
#pragma unroll
  for (int ky = 0; ky < 3; ++ky) {
#pragma unroll
    for (int kx = 0; kx < 3; ++kx) {
      int sy = y + ky - 1, sx = x + kx - 1;
      if (sy >= 0 && sy < H2 && sx >= 0 && sx < W2)
        msum += s[(b << 14) + (sy << 7) + sx];
    }
  }
  r[p]  = (msum > 0.f) ? ((float)K9 / fmaxf(msum, 1.f)) : 0.f;
  mh[p] = (msum > 0.f) ? 1.f : 0.f;
}

// ------------------------- WMMA helpers ------------------------------------
// A 16x32 bf16 fragment: lane L holds row M=L&15; K chunks at (half*8) and
// (16+half*8), each 8 contiguous bf16 (16B) -> packing matches VGPR layout.
// B 32x16 bf16 fragment: lane L holds column N=L&15, K chunks at (half*8) and
// (16+half*8), each 8 contiguous bf16, matching the A fragment K order.

__device__ __forceinline__ v16bf load_frag16(const bf16_t* p0, const bf16_t* p1) {
  v16bf f;
  *(v8bf*)&f       = *(const v8bf*)p0;
  *((v8bf*)&f + 1) = *(const v8bf*)p1;
  return f;
}

__device__ __forceinline__ v8f wmma_bf16(v16bf a, v16bf b, v8f c) {
  return __builtin_amdgcn_wmma_f32_16x16x32_bf16(false, a, false, b,
                                                 (short)0, c, false, false);
}

// ------------------------- expand GEMM: M=65536 N=768 K=1728 ---------------
// block = 8 waves; wave tile = 32M x 96N (acc[2][6]); grid (2048).
// M tile sits in one image row -> sy and row base are wave-uniform; only the
// horizontal halo bound is per-lane. EXEC is all-ones around every WMMA.

__global__ __launch_bounds__(256) void expand_wmma(
    const bf16_t* __restrict__ xm, const bf16_t* __restrict__ wB,
    const float* __restrict__ r, const float* __restrict__ mh,
    const float* __restrict__ bexp, bf16_t* __restrict__ hbuf) {
  int wave = threadIdx.x >> 5;
  int lane = threadIdx.x & 31;
  int half = lane >> 4;
  int l15  = lane & 15;
  int mt = blockIdx.x;               // 32-pixel M tile
  int n0 = wave * 96;
  int pbase = mt * 32;
  int b  = pbase >> 14;
  int y  = (pbase & 16383) >> 7;     // wave-uniform row
  int xb = pbase & 127;              // row-local start (0/32/64/96)
  int x0 = xb + l15;                 // subtile-0 column for this lane

  const bf16_t* brow[6];
#pragma unroll
  for (int j = 0; j < 6; ++j)
    brow[j] = wB + (size_t)(n0 + j * 16 + l15) * K9 + half * 8;

  v8bf zero8;
#pragma unroll
  for (int i = 0; i < 8; ++i) zero8[i] = (__bf16)0.0f;

  v8f acc[2][6] = {};
  for (int ky = 0; ky < 3; ++ky) {
    int sy = y + ky - 1;
    bool vy = (sy >= 0) & (sy < H2);              // wave-uniform
    const bf16_t* rowp = xm + ((size_t)(b * H2 + sy) * W2) * CH + half * 8;
    for (int kx = 0; kx < 3; ++kx) {
      int sx0 = x0 + kx - 1;
      int sx1 = sx0 + 16;
      bool v0 = vy & (sx0 >= 0) & (sx0 < W2);
      bool v1 = vy & (sx1 < W2);                  // sx1 >= 15 always
      const bf16_t* ap0 = rowp + (size_t)sx0 * CH;
      const bf16_t* ap1 = rowp + (size_t)sx1 * CH;
      int koff = (ky * 3 + kx) * CH;              // K offset of this tap
      for (int cb = 0; cb < 6; ++cb) {            // 6 x 32-wide K blocks
        v16bf a0, a1;
        if (v0) {
          const bf16_t* ap = ap0 + cb * 32;
          a0 = load_frag16(ap, ap + 16);
        } else {
          *(v8bf*)&a0 = zero8; *((v8bf*)&a0 + 1) = zero8;
        }
        if (v1) {
          const bf16_t* ap = ap1 + cb * 32;
          a1 = load_frag16(ap, ap + 16);
        } else {
          *(v8bf*)&a1 = zero8; *((v8bf*)&a1 + 1) = zero8;
        }
#pragma unroll
        for (int j = 0; j < 6; ++j) {
          const bf16_t* bp = brow[j] + koff + cb * 32;
          v16bf bb = load_frag16(bp, bp + 16);
          acc[0][j] = wmma_bf16(a0, bb, acc[0][j]);
          acc[1][j] = wmma_bf16(a1, bb, acc[1][j]);
        }
      }
    }
  }

  float rv[2][8], mv[2][8];
#pragma unroll
  for (int t = 0; t < 2; ++t)
#pragma unroll
    for (int v = 0; v < 8; ++v) {
      int pp = pbase + t * 16 + v + half * 8;   // D: M = v + 8*half
      rv[t][v] = r[pp];
      mv[t][v] = mh[pp];
    }
#pragma unroll
  for (int j = 0; j < 6; ++j) {
    int e = n0 + j * 16 + l15;
    float bias = bexp[e];
#pragma unroll
    for (int t = 0; t < 2; ++t)
#pragma unroll
      for (int v = 0; v < 8; ++v) {
        int pp = pbase + t * 16 + v + half * 8;
        float val = acc[t][j][v] * rv[t][v] + bias;
        float sv = val / (1.f + __expf(-val));               // SiLU
        hbuf[(size_t)pp * EX + e] = (bf16_t)(sv * mv[t][v]); // h*mh -> proj
      }
  }
}

// ------------------------- proj GEMM: M=65536 N=192 K=768 ------------------
// block = 4 waves; wave tile = 32M x 48N (acc[2][3]); grid (2048).

__global__ __launch_bounds__(128) void proj_wmma(
    const bf16_t* __restrict__ hbuf, const bf16_t* __restrict__ wP,
    const float* __restrict__ mh, const float* __restrict__ bproj,
    const bf16_t* __restrict__ xm, float* __restrict__ ym, bf16_t* __restrict__ z) {
  int wave = threadIdx.x >> 5;
  int lane = threadIdx.x & 31;
  int half = lane >> 4;
  int l15  = lane & 15;
  int mt = blockIdx.x;
  int n0 = wave * 48;
  int pbase = mt * 32;
  const bf16_t* arow0 = hbuf + (size_t)(pbase + l15) * EX + half * 8;
  const bf16_t* arow1 = arow0 + (size_t)16 * EX;
  const bf16_t* brow[3];
#pragma unroll
  for (int j = 0; j < 3; ++j)
    brow[j] = wP + (size_t)(n0 + j * 16 + l15) * EX + half * 8;

  v8f acc[2][3] = {};
  for (int kb = 0; kb < 24; ++kb) {
    const bf16_t* ap0 = arow0 + kb * 32;
    const bf16_t* ap1 = arow1 + kb * 32;
    v16bf a0 = load_frag16(ap0, ap0 + 16);
    v16bf a1 = load_frag16(ap1, ap1 + 16);
#pragma unroll
    for (int j = 0; j < 3; ++j) {
      const bf16_t* bp = brow[j] + kb * 32;
      v16bf bb = load_frag16(bp, bp + 16);
      acc[0][j] = wmma_bf16(a0, bb, acc[0][j]);
      acc[1][j] = wmma_bf16(a1, bb, acc[1][j]);
    }
  }

  float mv[2][8];
#pragma unroll
  for (int t = 0; t < 2; ++t)
#pragma unroll
    for (int v = 0; v < 8; ++v) mv[t][v] = mh[pbase + t * 16 + v + half * 8];
#pragma unroll
  for (int j = 0; j < 3; ++j) {
    int c = n0 + j * 16 + l15;
    float bias = bproj[c];
#pragma unroll
    for (int t = 0; t < 2; ++t)
#pragma unroll
      for (int v = 0; v < 8; ++v) {
        int pp = pbase + t * 16 + v + half * 8;
        float yv = acc[t][j][v] * mv[t][v] + bias + (float)xm[(size_t)pp * CH + c];
        ym[(size_t)pp * CH + c] = yv * mv[t][v];                // y*mask
        z[(size_t)pp * CH + c]  = (bf16_t)(yv * yv * mv[t][v]); // (y^2)*mask
      }
  }
}

// ------------------------- GDN GEMM: M=65536 N=192 K=192 -------------------
// norm = z @ gamma^T ; out = y*mask * rsqrt(beta + norm). Writes both tuple
// outputs (tensor + mask) in NCHW f32. block = 4 waves, wave = 32M x 48N.

__global__ __launch_bounds__(128) void gdn_wmma(
    const bf16_t* __restrict__ z, const bf16_t* __restrict__ gT,
    const float* __restrict__ ym, const float* __restrict__ mh,
    const float* __restrict__ beta, float* __restrict__ outp,
    float* __restrict__ maskp) {
  int wave = threadIdx.x >> 5;
  int lane = threadIdx.x & 31;
  int half = lane >> 4;
  int l15  = lane & 15;
  int mt = blockIdx.x;
  int n0 = wave * 48;
  int pbase = mt * 32;
  const bf16_t* arow0 = z + (size_t)(pbase + l15) * CH + half * 8;
  const bf16_t* arow1 = arow0 + (size_t)16 * CH;
  const bf16_t* brow[3];
#pragma unroll
  for (int j = 0; j < 3; ++j)
    brow[j] = gT + (size_t)(n0 + j * 16 + l15) * CH + half * 8;

  v8f acc[2][3] = {};
#pragma unroll
  for (int kb = 0; kb < 6; ++kb) {
    const bf16_t* ap0 = arow0 + kb * 32;
    const bf16_t* ap1 = arow1 + kb * 32;
    v16bf a0 = load_frag16(ap0, ap0 + 16);
    v16bf a1 = load_frag16(ap1, ap1 + 16);
#pragma unroll
    for (int j = 0; j < 3; ++j) {
      const bf16_t* bp = brow[j] + kb * 32;
      v16bf bb = load_frag16(bp, bp + 16);
      acc[0][j] = wmma_bf16(a0, bb, acc[0][j]);
      acc[1][j] = wmma_bf16(a1, bb, acc[1][j]);
    }
  }

  float mv[2][8];
#pragma unroll
  for (int t = 0; t < 2; ++t)
#pragma unroll
    for (int v = 0; v < 8; ++v) mv[t][v] = mh[pbase + t * 16 + v + half * 8];
#pragma unroll
  for (int j = 0; j < 3; ++j) {
    int i = n0 + j * 16 + l15;
    float be = beta[i];
#pragma unroll
    for (int t = 0; t < 2; ++t)
#pragma unroll
      for (int v = 0; v < 8; ++v) {
        int pp = pbase + t * 16 + v + half * 8;
        int rem = pp & 16383, bb_ = pp >> 14;
        float o = ym[(size_t)pp * CH + i] * rsqrtf(be + acc[t][j][v]);
        size_t oidx = (size_t)(bb_ * CH + i) * (H2 * W2) + rem;  // NCHW
        outp[oidx]  = o;
        maskp[oidx] = mv[t][v];
      }
  }
}

// ---------------------------------------------------------------------------

extern "C" void kernel_launch(void* const* d_in, const int* in_sizes, int n_in,
                              void* d_out, int out_size, void* d_ws, size_t ws_size,
                              hipStream_t stream) {
  (void)in_sizes; (void)n_in; (void)out_size; (void)ws_size;
  const float* ll     = (const float*)d_in[0];
  const float* b1     = (const float*)d_in[1];
  const float* b2     = (const float*)d_in[2];
  const float* b3     = (const float*)d_in[3];
  const int*   llm    = (const int*)d_in[4];
  const int*   m1     = (const int*)d_in[5];
  const int*   m2     = (const int*)d_in[6];
  const int*   m3     = (const int*)d_in[7];
  const float* w_cast = (const float*)d_in[8];
  const float* b_cast = (const float*)d_in[9];
  const float* w_exp  = (const float*)d_in[10];
  const float* b_exp  = (const float*)d_in[11];
  const float* w_proj = (const float*)d_in[12];
  const float* b_proj = (const float*)d_in[13];
  const float* gamma  = (const float*)d_in[14];
  const float* beta   = (const float*)d_in[15];

  char* ws = (char*)d_ws;
  size_t off = 0;
  auto alloc = [&](size_t bytes) -> char* {
    char* p = ws + off;
    off = (off + bytes + 255) & ~(size_t)255;
    return p;
  };
  float*  lo     = (float*) alloc((size_t)BN*CH*H2*HH*4);
  float*  hi     = (float*) alloc((size_t)BN*CH*H2*HH*4);
  float*  pre    = (float*) alloc((size_t)BN*CH*H2*W2*4);
  float*  mu     = (float*) alloc((size_t)BN*CH*HH*HH*4);
  bf16_t* xm     = (bf16_t*)alloc((size_t)BN*H2*W2*CH*2);
  bf16_t* mcast  = (bf16_t*)alloc((size_t)BN*H2*W2*CH*2);
  float*  s      = (float*) alloc((size_t)NPIX*4);
  float*  r      = (float*) alloc((size_t)NPIX*4);
  float*  mhv    = (float*) alloc((size_t)NPIX*4);
  bf16_t* hbuf   = (bf16_t*)alloc((size_t)BN*H2*W2*EX*2);
  float*  ym     = (float*) alloc((size_t)BN*H2*W2*CH*4);
  bf16_t* zb     = (bf16_t*)alloc((size_t)BN*H2*W2*CH*2);
  bf16_t* wExpT  = (bf16_t*)alloc((size_t)EX*K9*2);
  bf16_t* wProjT = (bf16_t*)alloc((size_t)CH*EX*2);
  bf16_t* gT     = (bf16_t*)alloc((size_t)CH*CH*2);

  float* outp  = (float*)d_out;
  float* maskp = outp + (size_t)BN*CH*H2*W2;

  const int T = 256;
  // weight repack
  prep_wexp<<<(EX*CH*9 + T-1)/T, T, 0, stream>>>(w_exp, wExpT);
  prep_copy_bf16<<<(CH*EX + T-1)/T, T, 0, stream>>>(w_proj, wProjT, CH*EX);
  prep_copy_bf16<<<(CH*CH + T-1)/T, T, 0, stream>>>(gamma, gT, CH*CH);
  // mask union + IDWT synthesis
  mu_kernel<<<(BN*CH*HH*HH + T-1)/T, T, 0, stream>>>(llm, m1, m2, m3, mu);
  idwt_vert<<<(BN*CH*H2*HH + T-1)/T, T, 0, stream>>>(ll, b1, b2, b3, llm, m1, m2, m3, lo, hi);
  idwt_horiz<<<(BN*CH*H2*W2 + T-1)/T, T, 0, stream>>>(lo, hi, pre);
  // depthwise cast partial conv -> NHWC bf16 activations + mask
  cast_pconv<<<(BN*H2*W2*CH + T-1)/T, T, 0, stream>>>(pre, mu, w_cast, b_cast, xm, mcast);
  // expand-conv renormalization terms
  chansum<<<(NPIX + T-1)/T, T, 0, stream>>>(mcast, s);
  renorm_expand<<<(NPIX + T-1)/T, T, 0, stream>>>(s, r, mhv);
  // heavy WMMA GEMMs (M tile = 32 pixels per block row)
  expand_wmma<<<dim3(NPIX/32), 256, 0, stream>>>(xm, wExpT, r, mhv, b_exp, hbuf);
  proj_wmma<<<dim3(NPIX/32), 128, 0, stream>>>(hbuf, wProjT, mhv, b_proj, xm, ym, zb);
  gdn_wmma<<<dim3(NPIX/32), 128, 0, stream>>>(zb, gT, ym, mhv, beta, outp, maskp);
}
